// NodePairRemovalDecoder_83820581749527
// MI455X (gfx1250) — compile-verified, matrix operand order checked by reading, not run on hardware
//
#include <hip/hip_runtime.h>

// ---------------------------------------------------------------------------
// Problem constants (from the reference)
// ---------------------------------------------------------------------------
#define NB     32          // batch
#define GP1    1001        // nodes per batch
#define HALFN  500
#define NH     8           // heads
#define DIM    512         // input dim (= NH * 64)
#define HD     64          // hidden per head
#define MROWS  (NB * GP1)  // 32032 GEMM rows (multiple of 16: 2002 tiles)
#define MTILES (MROWS / 16)
#define DIN    52          // 6*NH + 4 MLP input features

typedef __attribute__((ext_vector_type(16))) _Float16 v16h;
typedef __attribute__((ext_vector_type(8)))  float    v8f;
typedef __attribute__((ext_vector_type(4)))  unsigned int uix4;
typedef __attribute__((ext_vector_type(4)))  _Float16 h4;

union Frag { uix4 u[2]; v16h h; };

// ---------------------------------------------------------------------------
// h_hat f32 -> f16  (vectorized by 4)
// ---------------------------------------------------------------------------
__global__ void conv_h_k(const float* __restrict__ src, _Float16* __restrict__ dst, int n4) {
    int t = blockIdx.x * blockDim.x + threadIdx.x;
    if (t >= n4) return;
    float4 v = ((const float4*)src)[t];
    h4 o;
    o.x = (_Float16)v.x; o.y = (_Float16)v.y; o.z = (_Float16)v.z; o.w = (_Float16)v.w;
    ((h4*)dst)[t] = o;
}

// ---------------------------------------------------------------------------
// W [H,512,64] f32 -> Wt f16 laid out as B^T: Wt[c][d] = W[c/64][d][c%64]
// so B-fragment rows load contiguously (same pattern as A-fragments).
// ---------------------------------------------------------------------------
__global__ void conv_w_k(const float* __restrict__ w, _Float16* __restrict__ dst) {
    int t = blockIdx.x * blockDim.x + threadIdx.x;   // 512*512 per matrix
    if (t >= DIM * DIM) return;
    int c = t >> 9;          // output column (h*64+e)
    int d = t & 511;         // K index
    dst[t] = (_Float16)w[((c >> 6) << 15) + (d << 6) + (c & 63)];
}

// ---------------------------------------------------------------------------
// pre[b][sol[b][i]] = i   (inverse permutation)
// ---------------------------------------------------------------------------
__global__ void build_pre_k(const int* __restrict__ sol, int* __restrict__ pre) {
    int t = blockIdx.x * blockDim.x + threadIdx.x;
    if (t >= MROWS) return;
    int b = t / GP1;
    pre[b * GP1 + sol[t]] = t % GP1;
}

// post_post = sol[sol], pre_pre = pre[pre]
__global__ void compose_k(const int* __restrict__ sol, const int* __restrict__ pre,
                          int* __restrict__ postpost, int* __restrict__ prepre) {
    int t = blockIdx.x * blockDim.x + threadIdx.x;
    if (t >= MROWS) return;
    int b = t / GP1;
    postpost[t] = sol[b * GP1 + sol[t]];
    prepre[t]   = pre[b * GP1 + pre[t]];
}

// ---------------------------------------------------------------------------
// WMMA GEMM: Y[M x 512] = X[M x 512] * B, B given transposed as Wt[512c][512k].
// Block: 128 threads = 4 waves; each wave computes a 16x64 strip of C.
// grid = (512/64 = 8, ceil(MTILES/4)).
// Fragment layout per CDNA5 ISA (16-bit A 16x32; B loaded from B^T
// rows with the identical lane pattern; C/D: vgpr r -> row r or r+8).
// ---------------------------------------------------------------------------
__global__ void __launch_bounds__(128)
gemm16_k(const _Float16* __restrict__ X, const _Float16* __restrict__ Wt,
         float* __restrict__ Y, int M) {
    const int lane  = threadIdx.x & 31;
    const int wave  = threadIdx.x >> 5;
    const int mtile = blockIdx.y * 4 + wave;
    if (mtile >= (M + 15) / 16) return;
    const int row0 = mtile * 16;
    const int col0 = blockIdx.x * 64;
    const int rsel = lane & 15;
    const int ksel = (lane >> 4) << 3;   // lanes 16-31 hold K+8 / K+24

    v8f acc[4];
    #pragma unroll
    for (int i = 0; i < 4; ++i) acc[i] = (v8f){0.f,0.f,0.f,0.f,0.f,0.f,0.f,0.f};

    const _Float16* arow = X  + (size_t)(row0 + rsel) * DIM + ksel;
    const _Float16* brow = Wt + (size_t)(col0 + rsel) * DIM + ksel;

    for (int kb = 0; kb < DIM; kb += 32) {
        Frag a;
        a.u[0] = *(const uix4*)(arow + kb);        // K = kb+ksel   .. +7
        a.u[1] = *(const uix4*)(arow + kb + 16);   // K = kb+ksel+16.. +23
        #pragma unroll
        for (int ct = 0; ct < 4; ++ct) {
            const _Float16* bp = brow + (size_t)ct * 16 * DIM + kb;
            Frag b;
            b.u[0] = *(const uix4*)(bp);
            b.u[1] = *(const uix4*)(bp + 16);
            acc[ct] = __builtin_amdgcn_wmma_f32_16x16x32_f16(
                false, a.h, false, b.h, (short)0, acc[ct], false, false);
        }
    }

    const int rowOff = (lane >> 4) << 3;  // lanes 16-31 -> rows +8
    #pragma unroll
    for (int ct = 0; ct < 4; ++ct) {
        #pragma unroll
        for (int r = 0; r < 8; ++r) {
            Y[(size_t)(row0 + rowOff + r) * DIM + col0 + ct * 16 + (lane & 15)] = acc[ct][r];
        }
    }
}

// ---------------------------------------------------------------------------
// compat: val = Qg[idxA[j]].K[j] + Q[j].Kg[idxB[j]] - Qg[idxA[j]].Kg[idxB[j]]
// per (head, batch, node j=1..1000); scatter into feat[b*500+jj][colBase+half*8+h]
// Consecutive threads = consecutive heads -> coalesced 2KB row segments.
// ---------------------------------------------------------------------------
__global__ void compat_k(const float* __restrict__ Yq, const float* __restrict__ Yk,
                         const int* __restrict__ idxA, const int* __restrict__ idxB,
                         float* __restrict__ feat, int colBase) {
    int t = blockIdx.x * blockDim.x + threadIdx.x;
    if (t >= NH * 1000 * NB) return;
    int h  = t & 7;
    int j1 = (t >> 3) % 1000;
    int b  = t / (NH * 1000);
    int j  = j1 + 1;
    int base = b * GP1;
    int ia = idxA[base + j];
    int ib = idxB[base + j];
    const float4* qg = (const float4*)(Yq + (size_t)(base + ia) * DIM + h * HD);
    const float4* q  = (const float4*)(Yq + (size_t)(base + j ) * DIM + h * HD);
    const float4* kk = (const float4*)(Yk + (size_t)(base + j ) * DIM + h * HD);
    const float4* kg = (const float4*)(Yk + (size_t)(base + ib) * DIM + h * HD);
    float s1 = 0.f, s2 = 0.f, s3 = 0.f;
    #pragma unroll
    for (int e = 0; e < HD / 4; ++e) {
        float4 a = qg[e], c = kk[e], d = q[e], g = kg[e];
        s1 += a.x*c.x + a.y*c.y + a.z*c.z + a.w*c.w;
        s2 += d.x*g.x + d.y*g.y + d.z*g.z + d.w*g.w;
        s3 += a.x*g.x + a.y*g.y + a.z*g.z + a.w*g.w;
    }
    int half = (j1 >= HALFN) ? 1 : 0;
    int jj   = j1 - half * HALFN;
    feat[(size_t)(b * HALFN + jj) * DIN + colBase + half * 8 + h] = s1 + s2 - s3;
}

// selection_recent [B,4,500] -> feat cols 48..51
__global__ void sel_k(const float* __restrict__ sel, float* __restrict__ feat) {
    int t = blockIdx.x * blockDim.x + threadIdx.x;
    if (t >= NB * 4 * HALFN) return;
    int jj = t % HALFN;
    int s  = (t / HALFN) & 3;
    int b  = t / (4 * HALFN);
    feat[(size_t)(b * HALFN + jj) * DIN + 48 + s] = sel[t];
}

// ---------------------------------------------------------------------------
// Fused MLP: 52 -> 64 relu -> 32 relu -> 1. One thread per row, weights in LDS.
// ---------------------------------------------------------------------------
__global__ void __launch_bounds__(128)
mlp_k(const float* __restrict__ feat,
      const float* __restrict__ w1, const float* __restrict__ b1,
      const float* __restrict__ w2, const float* __restrict__ b2,
      const float* __restrict__ w3, const float* __restrict__ b3,
      float* __restrict__ out) {
    __shared__ float sw1[64 * DIN], sb1[64], sw2[32 * 64], sb2[32], sw3[32];
    for (int i = threadIdx.x; i < 64 * DIN; i += blockDim.x) sw1[i] = w1[i];
    for (int i = threadIdx.x; i < 32 * 64;  i += blockDim.x) sw2[i] = w2[i];
    if (threadIdx.x < 64) sb1[threadIdx.x] = b1[threadIdx.x];
    if (threadIdx.x < 32) { sb2[threadIdx.x] = b2[threadIdx.x]; sw3[threadIdx.x] = w3[threadIdx.x]; }
    __syncthreads();

    int t = blockIdx.x * blockDim.x + threadIdx.x;
    if (t >= NB * HALFN) return;

    float x[DIN];
    const float4* fr = (const float4*)(feat + (size_t)t * DIN);
    #pragma unroll
    for (int i = 0; i < DIN / 4; ++i) {
        float4 v = fr[i];
        x[4*i] = v.x; x[4*i+1] = v.y; x[4*i+2] = v.z; x[4*i+3] = v.w;
    }
    float h1[64];
    #pragma unroll
    for (int o = 0; o < 64; ++o) {
        float s = sb1[o];
        #pragma unroll
        for (int k = 0; k < DIN; ++k) s += sw1[o * DIN + k] * x[k];
        h1[o] = s > 0.f ? s : 0.f;
    }
    float ov = b3[0];
    #pragma unroll
    for (int o = 0; o < 32; ++o) {
        float s = sb2[o];
        #pragma unroll
        for (int k = 0; k < 64; ++k) s += sw2[o * 64 + k] * h1[k];
        ov += sw3[o] * (s > 0.f ? s : 0.f);
    }
    out[t] = ov;   // out is [B, 500] row-major == row index t
}

// ---------------------------------------------------------------------------
// Launch
// ---------------------------------------------------------------------------
extern "C" void kernel_launch(void* const* d_in, const int* in_sizes, int n_in,
                              void* d_out, int out_size, void* d_ws, size_t ws_size,
                              hipStream_t stream) {
    (void)in_sizes; (void)n_in; (void)out_size; (void)ws_size;

    const float* h_hat = (const float*)d_in[0];
    const int*   sol   = (const int*)  d_in[1];
    const float* selr  = (const float*)d_in[2];
    const float* Wsrc[6] = {(const float*)d_in[3], (const float*)d_in[4],
                            (const float*)d_in[5], (const float*)d_in[6],
                            (const float*)d_in[7], (const float*)d_in[8]};
    const float* fc1w = (const float*)d_in[9];
    const float* fc1b = (const float*)d_in[10];
    const float* fc2w = (const float*)d_in[11];
    const float* fc2b = (const float*)d_in[12];
    const float* fc3w = (const float*)d_in[13];
    const float* fc3b = (const float*)d_in[14];

    // Workspace layout (all 16B-aligned):
    char* ws = (char*)d_ws;
    constexpr size_t szX    = (size_t)MROWS * DIM * sizeof(_Float16);   // 32,800,768
    constexpr size_t szW    = 6ull * DIM * DIM * sizeof(_Float16);      //  3,145,728
    constexpr size_t szY    = (size_t)MROWS * DIM * sizeof(float);      // 65,601,536
    constexpr size_t szFeat = (size_t)NB * HALFN * DIN * sizeof(float); //  3,328,000
    constexpr size_t szIdx  = (size_t)MROWS * sizeof(int);

    _Float16* Xh   = (_Float16*)(ws);
    _Float16* Wt   = (_Float16*)(ws + szX);
    float*    Yq   = (float*)(ws + szX + szW);
    float*    Yk   = (float*)(ws + szX + szW + szY);
    float*    feat = (float*)(ws + szX + szW + 2 * szY);
    int*      pre  = (int*)  (ws + szX + szW + 2 * szY + szFeat);
    int*      prepre   = (int*)((char*)pre + szIdx);
    int*      postpost = (int*)((char*)pre + 2 * szIdx);

    // 1. conversions
    {
        int n4 = MROWS * DIM / 4;
        conv_h_k<<<(n4 + 255) / 256, 256, 0, stream>>>(h_hat, Xh, n4);
        for (int g = 0; g < 6; ++g)
            conv_w_k<<<(DIM * DIM) / 256, 256, 0, stream>>>(Wsrc[g], Wt + (size_t)g * DIM * DIM);
    }
    // 2. permutation indices
    build_pre_k<<<(MROWS + 255) / 256, 256, 0, stream>>>(sol, pre);
    compose_k  <<<(MROWS + 255) / 256, 256, 0, stream>>>(sol, pre, postpost, prepre);

    // 3. three compat groups: Q-GEMM, K-GEMM, gather-dot scatter into feat
    dim3 ggrid(DIM / 64, (MTILES + 3) / 4);
    for (int g = 0; g < 3; ++g) {
        gemm16_k<<<ggrid, 128, 0, stream>>>(Xh, Wt + (size_t)(2 * g)     * DIM * DIM, Yq, MROWS);
        gemm16_k<<<ggrid, 128, 0, stream>>>(Xh, Wt + (size_t)(2 * g + 1) * DIM * DIM, Yk, MROWS);
        const int* idxA = (g == 2) ? prepre : pre;
        const int* idxB = (g == 1) ? postpost : sol;
        compat_k<<<(NH * 1000 * NB) / 256, 256, 0, stream>>>(Yq, Yk, idxA, idxB, feat, g * 16);
    }

    // 4. selection features + fused MLP
    sel_k<<<(NB * 4 * HALFN + 255) / 256, 256, 0, stream>>>(selr, feat);
    mlp_k<<<(NB * HALFN) / 128, 128, 0, stream>>>(feat, fc1w, fc1b, fc2w, fc2b,
                                                  fc3w, fc3b, (float*)d_out);
}
